// ELLinkModel_85761906966871
// MI455X (gfx1250) — matrix operand order repaired and checked by reading, not compiled, not run on hardware
//
#include <hip/hip_runtime.h>

// ELLinkModel: embed -> LSTM x2 (bf16 WMMA, f32 accum) -> pool GEMM -> entity-max -> logits
// MI455X / gfx1250: wave32, v_wmma_f32_16x16x32_bf16, async global->LDS staging,
// 320KB LDS per WGP.

#define BDIM 64
#define SDIM 256
#define HDIM 768
#define H4   3072
#define KTOT 1536           // concat [x|h] K dimension
#define KT   48             // KTOT/32 k-tiles per step
#define NT_H 48             // HDIM/16 hidden-column tiles
#define TDIM 2
#define KPOOL (SDIM*HDIM)   // 196608
#define KSPLIT 64
#define KT_PER 96           // (KPOOL/32)/KSPLIT

#if defined(__AMDGCN__) && __has_builtin(__builtin_amdgcn_global_load_async_to_lds_b128)
#define ASYNC_LDS 1
#else
#define ASYNC_LDS 0
#endif

typedef __attribute__((ext_vector_type(16))) __bf16 v16bf;
typedef __attribute__((ext_vector_type(8)))  float  v8f;
typedef int v4i __attribute__((__vector_size__(16)));

#if ASYNC_LDS
typedef __attribute__((address_space(1))) v4i* gvec128_t;
typedef __attribute__((address_space(3))) v4i* lvec128_t;
#endif

union Frag {
    v16bf v;
    unsigned u[8];
    uint4 q[2];
};

__device__ __forceinline__ unsigned short f2bf(float f) {
    unsigned u = __float_as_uint(f);
    u += 0x7FFFu + ((u >> 16) & 1u);            // round-to-nearest-even
    return (unsigned short)(u >> 16);
}
__device__ __forceinline__ float bf2f(unsigned short h) {
    return __uint_as_float(((unsigned)h) << 16);
}
__device__ __forceinline__ float sigm(float x) { return 1.0f / (1.0f + __expf(-x)); }

// 16-byte global -> LDS copy: async-to-LDS path (no VGPR round trip) when available.
__device__ __forceinline__ void cp16(unsigned short* ldsDst, const unsigned short* gSrc) {
#if ASYNC_LDS
    __builtin_amdgcn_global_load_async_to_lds_b128(
        (gvec128_t)(void*)gSrc, (lvec128_t)(void*)ldsDst, 0, 0);
#else
    *(uint4*)ldsDst = *(const uint4*)gSrc;
#endif
}

__device__ __forceinline__ void stage_fence() {
#if ASYNC_LDS
#if __has_builtin(__builtin_amdgcn_s_wait_asynccnt)
    __builtin_amdgcn_s_wait_asynccnt(0);
#else
    asm volatile("s_wait_asynccnt 0x0" ::: "memory");
#endif
#endif
    __syncthreads();
}

// ---------------- embedding gather -> bf16 [B][S][H] ----------------
__global__ void k_embed(const int* __restrict__ ids, const float* __restrict__ table,
                        unsigned short* __restrict__ out) {
    size_t gid = (size_t)blockIdx.x * 256 + threadIdx.x;   // < B*S*H
    int bs = (int)(gid / HDIM);
    int h  = (int)(gid % HDIM);
    int id = ids[bs];
    out[gid] = f2bf(table[(size_t)id * HDIM + h]);
}

// ---------------- pack [Wx;Wh] f32 -> WMMA-B bf16 tile layout ----------------
// Wp dword index = ((nt*KT + kt)*32 + lane)*8 + v
// lane<16: N=lane, K_local = 2v,2v+1 ; lane>=16: N=lane-16, K_local = 16+2v,17+2v
__global__ void k_pack_w(const float* __restrict__ Wx, const float* __restrict__ Wh,
                         unsigned* __restrict__ Wp) {
    int gid = blockIdx.x * 256 + threadIdx.x;    // < 192*48*256 = 2359296
    int v    = gid & 7;
    int lane = (gid >> 3) & 31;
    int tile = gid >> 8;
    int kt = tile % KT;
    int nt = tile / KT;                          // 0..191 (gate*48 + hidden-block)
    int n  = nt * 16 + (lane & 15);
    int k0 = kt * 32 + ((lane >> 4) << 4) + (v << 1);
    float f0 = (k0 < HDIM) ? Wx[(size_t)k0 * H4 + n] : Wh[(size_t)(k0 - HDIM) * H4 + n];
    int k1 = k0 + 1;
    float f1 = (k1 < HDIM) ? Wx[(size_t)k1 * H4 + n] : Wh[(size_t)(k1 - HDIM) * H4 + n];
    Wp[gid] = (unsigned)f2bf(f0) | ((unsigned)f2bf(f1) << 16);
}

// ---------------- zero h0 (bf16) and c (f32) ----------------
__global__ void k_zero(unsigned short* __restrict__ h0, float* __restrict__ c) {
    int i = blockIdx.x * 256 + threadIdx.x;      // < B*H = 49152
    h0[i] = 0;
    c[i]  = 0.0f;
}

// ---------------- one LSTM timestep ----------------
// grid 48 (hidden 16-col slices), block 256 (8 waves). Dynamic LDS:
//   A = [64][1536] bf16 (192KB) ; gates = [4][64][16] f32 (16KB)
__global__ void __launch_bounds__(256) k_lstm_step(
    const unsigned short* __restrict__ xsrc,   // [B][S][H] bf16 bits
    const unsigned short* __restrict__ hprev,  // [B][H]
    unsigned short* __restrict__ hnext,        // [B][H]
    float* __restrict__ cstate,                // [B][H]
    const unsigned* __restrict__ Wp,           // packed [192][48][256] dwords
    const float* __restrict__ bias,            // [4H]
    unsigned short* __restrict__ outseq,       // [B][S][H]
    int t)
{
    extern __shared__ char smem[];
    unsigned short* Alds = (unsigned short*)smem;                         // [64][1536]
    float* gl = (float*)(smem + (size_t)64 * KTOT * sizeof(unsigned short)); // [4][64][16]

    const int tid = threadIdx.x;

    // stage A = [x_t | h_prev] rows; 16B chunks, async global->LDS when available
    for (int i = tid; i < 64 * 96; i += 256) {
        int r = i / 96, c = i % 96;
        cp16(Alds + r * KTOT + c * 8,
             xsrc + ((size_t)r * SDIM + t) * HDIM + c * 8);
    }
    for (int i = tid; i < 64 * 96; i += 256) {
        int r = i / 96, c = i % 96;
        cp16(Alds + r * KTOT + HDIM + c * 8,
             hprev + (size_t)r * HDIM + c * 8);
    }
    stage_fence();

    const int lane = tid & 31;
    const int w    = tid >> 5;          // wave id (wave32)
    const int g    = w & 3;             // gate (i,f,g,o)
    const int mt0  = (w >> 2) * 2;      // batch-tile pair
    const int mt1  = mt0 + 1;
    const int hi   = lane >> 4;
    const int lm   = lane & 15;
    const int ntg  = g * NT_H + blockIdx.x;   // global 16-col tile in [0,192)

    v8f acc0 = {}, acc1 = {};
    // A layout per lane: K = hi*16 + 2v,2v+1 -> one contiguous 32B run (two 16B groups)
    const unsigned short* base0 = Alds + (size_t)(mt0 * 16 + lm) * KTOT + (hi << 4);
    const unsigned short* base1 = Alds + (size_t)(mt1 * 16 + lm) * KTOT + (hi << 4);
    for (int kk = 0; kk < KT; ++kk) {
        Frag b;
        const uint4* bp = (const uint4*)(Wp + (((size_t)ntg * KT + kk) << 8) + (lane << 3));
        b.q[0] = bp[0];
        b.q[1] = bp[1];
        Frag a0, a1;
        const unsigned short* r0 = base0 + kk * 32;
        const unsigned short* r1 = base1 + kk * 32;
        a0.q[0] = *(const uint4*)(r0);          // K group 0..7   (ds_load_b128)
        a0.q[1] = *(const uint4*)(r0 + 8);      // K group 8..15  (ds_load_b128)
        a1.q[0] = *(const uint4*)(r1);
        a1.q[1] = *(const uint4*)(r1 + 8);
        acc0 = __builtin_amdgcn_wmma_f32_16x16x32_bf16(false, a0.v, false, b.v,
                                                       (short)0, acc0, false, false);
        acc1 = __builtin_amdgcn_wmma_f32_16x16x32_bf16(false, a1.v, false, b.v,
                                                       (short)0, acc1, false, false);
    }

    // D layout: lane L, vgpr r -> (M = r + 8*(L>>4), N = L&15)
    #pragma unroll
    for (int r = 0; r < 8; ++r) {
        gl[(g * 64 + mt0 * 16 + r + hi * 8) * 16 + lm] = acc0[r];
        gl[(g * 64 + mt1 * 16 + r + hi * 8) * 16 + lm] = acc1[r];
    }
    __syncthreads();

    // elementwise LSTM update for this WG's 16 hidden columns
    const int jbase = blockIdx.x * 16;
    #pragma unroll
    for (int it = 0; it < 4; ++it) {
        int idx = it * 256 + tid;    // 0..1023 = 64 batch x 16 cols
        int m = idx >> 4, j = idx & 15;
        int jg = jbase + j;
        float gi = gl[(0 * 64 + m) * 16 + j] + bias[jg];
        float gf = gl[(1 * 64 + m) * 16 + j] + bias[HDIM + jg];
        float gg = gl[(2 * 64 + m) * 16 + j] + bias[2 * HDIM + jg];
        float go = gl[(3 * 64 + m) * 16 + j] + bias[3 * HDIM + jg];
        size_t ci = (size_t)m * HDIM + jg;
        float c  = cstate[ci];
        float cn = sigm(gf) * c + sigm(gi) * tanhf(gg);
        float hn = sigm(go) * tanhf(cn);
        cstate[ci] = cn;
        unsigned short hb = f2bf(hn);
        hnext[ci] = hb;
        outseq[((size_t)m * SDIM + t) * HDIM + jg] = hb;
    }
}

// ---------------- pool GEMM: [64,196608]bf16 @ [196608,768]f32 -> K-split partials ----
__global__ void __launch_bounds__(128) k_pool_gemm(
    const unsigned short* __restrict__ Abf,   // [64][196608] bf16 bits
    const float* __restrict__ Wpool,          // [196608][768] f32
    float* __restrict__ partial)              // [KSPLIT][64][768] f32
{
    const int tid  = threadIdx.x;
    const int lane = tid & 31;
    const int mt   = tid >> 5;      // 4 waves = 4 batch tiles
    const int hi   = lane >> 4;
    const int lm   = lane & 15;
    const int nt   = blockIdx.x;    // 0..47
    const int ks   = blockIdx.y;    // 0..63
    const int n    = nt * 16 + lm;

    v8f acc = {};
    const unsigned short* rowp = Abf + (size_t)(mt * 16 + lm) * KPOOL + (hi << 4);
    for (int kt = ks * KT_PER; kt < (ks + 1) * KT_PER; ++kt) {
        int K0 = kt * 32;
        Frag a, b;
        a.q[0] = *(const uint4*)(rowp + K0);
        a.q[1] = *(const uint4*)(rowp + K0 + 8);
        int kb = K0 + (hi << 4);
        #pragma unroll
        for (int v = 0; v < 8; ++v) {
            float f0 = Wpool[(size_t)(kb + 2 * v) * HDIM + n];
            float f1 = Wpool[(size_t)(kb + 2 * v + 1) * HDIM + n];
            b.u[v] = (unsigned)f2bf(f0) | ((unsigned)f2bf(f1) << 16);
        }
        acc = __builtin_amdgcn_wmma_f32_16x16x32_bf16(false, a.v, false, b.v,
                                                      (short)0, acc, false, false);
    }
    #pragma unroll
    for (int r = 0; r < 8; ++r) {
        partial[((size_t)ks * 64 + mt * 16 + r + hi * 8) * HDIM + nt * 16 + lm] = acc[r];
    }
}

// ---------------- reduce K-split partials + pool bias ----------------
__global__ void k_pool_reduce(const float* __restrict__ partial,
                              const float* __restrict__ pb,
                              float* __restrict__ poolout) {
    int gid = blockIdx.x * 256 + threadIdx.x;   // < 49152
    int j = gid % HDIM;
    float s = pb[j];
    for (int ks = 0; ks < KSPLIT; ++ks) s += partial[(size_t)ks * 64 * HDIM + gid];
    poolout[gid] = s;
}

// ---------------- entity-max, max with pool, logits ----------------
__global__ void k_final(const unsigned short* __restrict__ out1,
                        const int* __restrict__ entity,
                        const float* __restrict__ poolout,
                        const float* __restrict__ linW,
                        const float* __restrict__ linb,
                        float* __restrict__ logits) {
    __shared__ float pooled[HDIM];
    __shared__ float red[256];
    const int b = blockIdx.x, tid = threadIdx.x;
    for (int j = tid; j < HDIM; j += 256) {
        float m = -3.402823466e+38f;
        const unsigned short* base = out1 + (size_t)b * SDIM * HDIM + j;
        for (int s = 0; s < SDIM; ++s) {
            float v = (entity[b * SDIM + s] == 1) ? bf2f(base[(size_t)s * HDIM]) : 0.0f;
            m = fmaxf(m, v);
        }
        pooled[j] = fmaxf(m, poolout[(size_t)b * HDIM + j]);
    }
    __syncthreads();
    for (int tc = 0; tc < TDIM; ++tc) {
        float p = 0.0f;
        for (int j = tid; j < HDIM; j += 256) p += pooled[j] * linW[j * TDIM + tc];
        red[tid] = p;
        __syncthreads();
        for (int off = 128; off > 0; off >>= 1) {
            if (tid < off) red[tid] += red[tid + off];
            __syncthreads();
        }
        if (tid == 0) logits[b * TDIM + tc] = red[0] + linb[tc];
        __syncthreads();
    }
}

extern "C" void kernel_launch(void* const* d_in, const int* in_sizes, int n_in,
                              void* d_out, int out_size, void* d_ws, size_t ws_size,
                              hipStream_t stream) {
    const int*   ids    = (const int*)d_in[0];
    const int*   entity = (const int*)d_in[4];
    const float* table  = (const float*)d_in[5];
    const float* Wx0    = (const float*)d_in[6];
    const float* Wh0    = (const float*)d_in[7];
    const float* b0     = (const float*)d_in[8];
    const float* Wx1    = (const float*)d_in[9];
    const float* Wh1    = (const float*)d_in[10];
    const float* b1     = (const float*)d_in[11];
    const float* poolW  = (const float*)d_in[12];
    const float* poolb  = (const float*)d_in[13];
    const float* linW   = (const float*)d_in[14];
    const float* linb   = (const float*)d_in[15];
    float* logits = (float*)d_out;

    char* p = (char*)d_ws;
    auto carve = [&](size_t bytes) { char* r = p; p += (bytes + 255) & ~(size_t)255; return r; };
    unsigned short* emb  = (unsigned short*)carve((size_t)BDIM * SDIM * HDIM * 2); // 24MB
    unsigned short* out0 = (unsigned short*)carve((size_t)BDIM * SDIM * HDIM * 2); // 24MB
    unsigned short* out1 = (unsigned short*)carve((size_t)BDIM * SDIM * HDIM * 2); // 24MB
    unsigned*       Wp0  = (unsigned*)carve((size_t)192 * 48 * 256 * 4);           // 9MB
    unsigned*       Wp1  = (unsigned*)carve((size_t)192 * 48 * 256 * 4);           // 9MB
    unsigned short* hb0  = (unsigned short*)carve((size_t)BDIM * HDIM * 2);
    unsigned short* hb1  = (unsigned short*)carve((size_t)BDIM * HDIM * 2);
    float*          cst  = (float*)carve((size_t)BDIM * HDIM * 4);
    float*          part = (float*)carve((size_t)KSPLIT * BDIM * HDIM * 4);        // 12.6MB
    float*          pout = (float*)carve((size_t)BDIM * HDIM * 4);

    const size_t smem = (size_t)64 * KTOT * 2 + (size_t)4 * 64 * 16 * 4;  // 208KB
    (void)hipFuncSetAttribute((const void*)k_lstm_step,
                              hipFuncAttributeMaxDynamicSharedMemorySize, (int)smem);

    k_embed<<<(BDIM * SDIM * HDIM) / 256, 256, 0, stream>>>(ids, table, emb);
    k_pack_w<<<(192 * 48 * 256) / 256, 256, 0, stream>>>(Wx0, Wh0, Wp0);
    k_pack_w<<<(192 * 48 * 256) / 256, 256, 0, stream>>>(Wx1, Wh1, Wp1);

    unsigned short* hb[2] = {hb0, hb1};

    k_zero<<<(BDIM * HDIM) / 256, 256, 0, stream>>>(hb0, cst);
    for (int t = 0; t < SDIM; ++t)
        k_lstm_step<<<NT_H, 256, smem, stream>>>(emb, hb[t & 1], hb[(t + 1) & 1],
                                                 cst, Wp0, b0, out0, t);

    k_zero<<<(BDIM * HDIM) / 256, 256, 0, stream>>>(hb0, cst);
    for (int t = 0; t < SDIM; ++t)
        k_lstm_step<<<NT_H, 256, smem, stream>>>(out0, hb[t & 1], hb[(t + 1) & 1],
                                                 cst, Wp1, b1, out1, t);

    k_pool_gemm<<<dim3(NT_H, KSPLIT), 128, 0, stream>>>(out1, poolW, part);
    k_pool_reduce<<<(BDIM * HDIM) / 256, 256, 0, stream>>>(part, poolb, pout);
    k_final<<<BDIM, 256, 0, stream>>>(out1, entity, pout, linW, linb, logits);

    (void)in_sizes; (void)n_in; (void)out_size; (void)ws_size;
}
